// Decoder_12627203851074
// MI455X (gfx1250) — matrix operand order repaired and checked by reading, not compiled, not run on hardware
//
#include <hip/hip_runtime.h>

#define B_  32
#define P_  2048
#define S_  1024
#define H_  256
#define NP_ (B_*P_)   // 65536 goal rows
#define NS_ (B_*S_)   // 32768 hidden rows

typedef __attribute__((ext_vector_type(16))) __bf16 v16bf;
typedef __attribute__((ext_vector_type(8)))  __bf16 v8bf;
typedef __attribute__((ext_vector_type(8)))  float  v8f;
typedef __attribute__((ext_vector_type(4)))  unsigned int u32x4;
typedef __attribute__((ext_vector_type(8)))  unsigned int u32x8;

// ---------------------------------------------------------------- wave utils
__device__ __forceinline__ float wsum32(float v) {
#pragma unroll
  for (int o = 16; o > 0; o >>= 1) v += __shfl_xor(v, o, 32);
  return v;
}
__device__ __forceinline__ float wmax32(float v) {
#pragma unroll
  for (int o = 16; o > 0; o >>= 1) v = fmaxf(v, __shfl_xor(v, o, 32));
  return v;
}
// reduce within each 16-lane half of the wave
__device__ __forceinline__ float hsum16(float v) {
#pragma unroll
  for (int o = 1; o < 16; o <<= 1) v += __shfl_xor(v, o, 32);
  return v;
}
__device__ __forceinline__ float hmax16(float v) {
#pragma unroll
  for (int o = 1; o < 16; o <<= 1) v = fmaxf(v, __shfl_xor(v, o, 32));
  return v;
}

// ---------------------------------------------------------------- WMMA helpers
__device__ __forceinline__ v8f wmma_bf16(v16bf a, v16bf b, v8f c) {
  return __builtin_amdgcn_wmma_f32_16x16x32_bf16(
      /*neg_a=*/false, a, /*neg_b=*/false, b,
      /*c_mod=*/(short)0, c, /*reuse_a=*/false, /*reuse_b=*/false);
}

// A fragment: 16x32 bf16 tile at `base` (row-major, ld elements/row), K offset kb.
__device__ __forceinline__ v16bf load_fragA(const __bf16* base, int ld, int kb) {
  const int l  = threadIdx.x & 31;
  const int m  = l & 15;
  const int ks = (l < 16) ? 0 : 8;
  const __bf16* p = base + m * ld + kb + ks;
  v8bf lo = *(const v8bf*)(p);
  v8bf hi = *(const v8bf*)(p + 16);
  v16bf r;
#pragma unroll
  for (int i = 0; i < 8; ++i) { r[i] = lo[i]; r[8 + i] = hi[i]; }
  return r;
}

// B fragment (32x16): B[k][n] = Wsrc[nbase+n][k] (Wsrc row-major, ld elems/row).
__device__ __forceinline__ v16bf load_fragB(const __bf16* Wsrc, int ld, int nbase, int kb) {
  const int l  = threadIdx.x & 31;
  const int n  = nbase + (l & 15);
  const int ks = (l < 16) ? 0 : 8;
  const __bf16* p = Wsrc + (long)n * ld + kb + ks;
  v8bf lo = *(const v8bf*)(p);
  v8bf hi = *(const v8bf*)(p + 16);
  v16bf r;
#pragma unroll
  for (int i = 0; i < 8; ++i) { r[i] = lo[i]; r[8 + i] = hi[i]; }
  return r;
}

// --------------------------------------------------- TDM: 16x32 bf16 tile -> LDS
// D# group0/group1 built in SGPRs; groups 2/3 zero (2-D tensor).
__device__ __forceinline__ void tdm_load_tile16x32(const __bf16* ldsDst,
                                                   const __bf16* gsrc, int lda) {
  unsigned lds = (unsigned)(unsigned long long)ldsDst;       // LDS byte address
  unsigned long long ga = (unsigned long long)gsrc;          // tile start
  u32x4 g0;
  g0[0] = 1u;                                                // count=1, no gather
  g0[1] = lds;                                               // lds_addr
  g0[2] = (unsigned)ga;                                      // global_addr[31:0]
  g0[3] = (unsigned)((ga >> 32) & 0x1FFFFFFull) | (2u << 30);// addr[56:32] | type=2
  u32x8 g1;
  g1[0] = (1u << 16);                                        // data_size=2B
  g1[1] = ((unsigned)lda & 0xFFFFu) << 16;                   // tensor_dim0 lo16
  g1[2] = ((unsigned)lda >> 16) | (16u << 16);               // dim0 hi | tensor_dim1=16
  g1[3] = (32u << 16);                                       // tile_dim0=32
  g1[4] = 16u;                                               // tile_dim1=16, tile_dim2=0
  g1[5] = (unsigned)lda;                                     // tensor_dim0_stride lo32
  g1[6] = 0u;
  g1[7] = 0u;
  u32x4 gz; gz[0] = 0u; gz[1] = 0u; gz[2] = 0u; gz[3] = 0u;
  asm volatile("tensor_load_to_lds %0, %1, %2, %3"
               :: "s"(g0), "s"(g1), "s"(gz), "s"(gz)
               : "memory");
}

// ---------------------------------------------------------------- cast fp32->bf16
__global__ void cast_f32_bf16(const float* __restrict__ src, __bf16* __restrict__ dst, int n) {
  int i = blockIdx.x * blockDim.x + threadIdx.x;
  if (i < n) dst[i] = (__bf16)src[i];
}

// ------------------------------------------- K=2 head MLP: relu(LN(x@W^T+b))
__global__ void head_mlp(const float* __restrict__ X, const float* __restrict__ W,
                         const float* __restrict__ bias, const float* __restrict__ g,
                         const float* __restrict__ be, __bf16* __restrict__ Y,
                         int ldy, int Mout, int Nrows) {
  const int wave = threadIdx.x >> 5, lane = threadIdx.x & 31;
  const int row = blockIdx.x * 8 + wave;
  if (row >= Nrows) return;
  const float x0 = X[row * 2 + 0], x1 = X[row * 2 + 1];
  const int per = Mout / 32;
  float vals[8];
  float s = 0.f, ss = 0.f;
  for (int i = 0; i < per; ++i) {
    int j = lane + i * 32;
    float v = W[j * 2 + 0] * x0 + W[j * 2 + 1] * x1 + bias[j];
    vals[i] = v; s += v; ss += v * v;
  }
  s = wsum32(s); ss = wsum32(ss);
  const float mean = s / (float)Mout;
  const float inv  = rsqrtf(ss / (float)Mout - mean * mean + 1e-5f);
  for (int i = 0; i < per; ++i) {
    int j = lane + i * 32;
    float v = (vals[i] - mean) * inv * g[j] + be[j];
    Y[(long)row * ldy + j] = (__bf16)fmaxf(v, 0.f);
  }
}

// ---------------------- broadcast agent=hidden[:,0,:] into cat4 / x1 / x2 halves
__global__ void fill_agent(const float* __restrict__ hidden, __bf16* __restrict__ cat4,
                           __bf16* __restrict__ x1, __bf16* __restrict__ x2) {
  const long i = (long)blockIdx.x * blockDim.x + threadIdx.x;  // B*P*H threads
  const int h = (int)(i % H_);
  const long bp = i / H_;
  const int b = (int)(bp / P_);
  const float v = hidden[((long)b * S_ + 0) * H_ + h];
  cat4[bp * 1024 + h] = (__bf16)v;
  if (h < 128) {
    x1[bp * 256 + 128 + h] = (__bf16)v;
    x2[bp * 256 + 128 + h] = (__bf16)v;
  }
}

// ---------------------------------------------------------------- WMMA GEMM
// Y(N x Mout) = X(N x K) @ W(Mout x K)^T (+bias)
// MODE 0: Y=(XW^T+b)*scale (bf16)   MODE 1: Y=relu(LN(XW^T+b))
// MODE 2: transposed per batch of Srows: Yt[b][n][s]
// TPW = column tiles per wave (Mout = TPW*128). A tiles double-buffered via TDM.
template<int TPW, int MODE>
__global__ void gemm_bf16(const __bf16* __restrict__ A, int lda,
                          const __bf16* __restrict__ W, int ldw,
                          const float* __restrict__ bias,
                          const float* __restrict__ gam,
                          const float* __restrict__ bet,
                          __bf16* __restrict__ Y, int ldy,
                          int K, int Mout, float scale, int Srows) {
  __shared__ __align__(16) __bf16 Atile[2][16 * 32];
  __shared__ __align__(16) float  Ytile[16 * 256];

  const int wave = threadIdx.x >> 5, lane = threadIdx.x & 31;
  const int rowBase = blockIdx.x * 16;
  const __bf16* Arow = A + (long)rowBase * lda;

  v8f zero = {0.f, 0.f, 0.f, 0.f, 0.f, 0.f, 0.f, 0.f};
  v8f acc[TPW];
#pragma unroll
  for (int t = 0; t < TPW; ++t) acc[t] = zero;

  __builtin_prefetch((const void*)(W + (long)(wave * 16) * ldw), 0, 1);

  const int nsteps = K >> 5;
  if (wave == 0) tdm_load_tile16x32(&Atile[0][0], Arow, lda);

  for (int i = 0; i < nsteps; ++i) {
    const int cur = i & 1;
    if (wave == 0) {
      if (i + 1 < nsteps) {
        tdm_load_tile16x32(&Atile[1 - cur][0], Arow + (i + 1) * 32, lda);
        __builtin_amdgcn_s_wait_tensorcnt(1);   // wait for tile i, keep i+1 in flight
      } else {
        __builtin_amdgcn_s_wait_tensorcnt(0);
      }
    }
    __syncthreads();                            // Atile[cur] visible to all waves
    v16bf af = load_fragA(&Atile[cur][0], 32, 0);
#pragma unroll
    for (int t = 0; t < TPW; ++t) {
      const int nb = (wave + t * 8) * 16;
      v16bf bf = load_fragB(W, ldw, nb, i * 32);
      acc[t] = wmma_bf16(af, bf, acc[t]);
    }
    __syncthreads();                            // done reading before TDM overwrites
  }

  // epilogue
  const int mo = (lane < 16) ? 0 : 8;
#pragma unroll
  for (int t = 0; t < TPW; ++t) {
    const int nb = (wave + t * 8) * 16;
    const int n  = nb + (lane & 15);
    const float bv = bias[n];
#pragma unroll
    for (int i = 0; i < 8; ++i) {
      float v = (acc[t][i] + bv) * scale;
      if (MODE == 1) {
        Ytile[(i + mo) * Mout + n] = v;
      } else if (MODE == 0) {
        Y[(long)(rowBase + i + mo) * ldy + n] = (__bf16)v;
      } else {  // MODE 2: transposed per batch
        int row = rowBase + i + mo;
        int bb = row / Srows, s2 = row - bb * Srows;
        Y[((long)bb * Mout + n) * Srows + s2] = (__bf16)v;
      }
    }
  }

  if (MODE == 1) {
    __syncthreads();
    const int per = Mout / 32;
#pragma unroll
    for (int rr = 0; rr < 2; ++rr) {
      const int m = wave * 2 + rr;
      float s = 0.f, ss = 0.f;
      float vv[8];
      for (int i = 0; i < per; ++i) {
        int j = lane + i * 32;
        float v = Ytile[m * Mout + j];
        vv[i] = v; s += v; ss += v * v;
      }
      s = wsum32(s); ss = wsum32(ss);
      const float mean = s / (float)Mout;
      const float inv  = rsqrtf(ss / (float)Mout - mean * mean + 1e-5f);
      for (int i = 0; i < per; ++i) {
        int j = lane + i * 32;
        float v = (vv[i] - mean) * inv * gam[j] + bet[j];
        Y[(long)(rowBase + m) * ldy + j] = (__bf16)fmaxf(v, 0.f);
      }
    }
  }
}

// ---------------------------------------------------------------- attention
__global__ void attention_kernel(const __bf16* __restrict__ Q,   // (B*P, H) q/sqrt(H)
                                 const __bf16* __restrict__ Km,  // (B*S, H)
                                 const __bf16* __restrict__ Vt,  // (B, H, S)
                                 const float*  __restrict__ mask,// (B, P, S)
                                 __bf16* __restrict__ cat4) {    // attn -> +768
  __shared__ __align__(16) __bf16 qt[16 * H_];     // 8 KB
  __shared__ __align__(16) __bf16 pr[16 * 1024];   // 32 KB
  __shared__ float pmax[16 * 8];
  __shared__ float psum[16 * 8];

  const int wave = threadIdx.x >> 5, lane = threadIdx.x & 31;
  const int b  = blockIdx.x >> 7;
  const int pt = blockIdx.x & 127;
  const int p0 = pt * 16;
  const int mo = (lane < 16) ? 0 : 8;

  // async copy of the contiguous 16x256 bf16 q tile into LDS (ASYNCcnt path)
  {
    const __bf16* qbase = Q + (long)(b * P_ + p0) * H_;
    unsigned ldsq = (unsigned)(unsigned long long)&qt[0];
#pragma unroll
    for (int it = 0; it < 2; ++it) {
      const int off = (threadIdx.x + it * 256) * 16;   // bytes
      unsigned d = ldsq + (unsigned)off;
      const char* sp = (const char*)qbase + off;
      asm volatile("global_load_async_to_lds_b128 %0, %1, off"
                   :: "v"(d), "v"(sp) : "memory");
    }
    asm volatile("s_wait_asynccnt 0x0" ::: "memory");
  }
  __syncthreads();

  const __bf16* Kbase = Km + (long)b * S_ * H_;
  v8f zero = {0.f, 0.f, 0.f, 0.f, 0.f, 0.f, 0.f, 0.f};
  v8f acc[8];

  // phase 1: scores (each wave owns 8 s-tiles, fp32 in registers)
#pragma unroll
  for (int j = 0; j < 8; ++j) {
    const int st = wave + j * 8;
    v8f a = zero;
    for (int kb = 0; kb < H_; kb += 32) {
      v16bf af = load_fragA(qt, H_, kb);
      v16bf bf = load_fragB(Kbase, H_, st * 16, kb);
      a = wmma_bf16(af, bf, a);
    }
    const int s = st * 16 + (lane & 15);
#pragma unroll
    for (int i = 0; i < 8; ++i) {
      float msk = mask[(long)(b * P_ + p0 + i + mo) * S_ + s];
      a[i] += (1.0f - msk) * (-10000.0f);
    }
    acc[j] = a;
  }

  // phase 2: softmax over S
  float rmax[8], rsum[8];
#pragma unroll
  for (int i = 0; i < 8; ++i) {
    float mx = acc[0][i];
#pragma unroll
    for (int j = 1; j < 8; ++j) mx = fmaxf(mx, acc[j][i]);
    rmax[i] = hmax16(mx);
  }
  if ((lane & 15) == 0) {
#pragma unroll
    for (int i = 0; i < 8; ++i) pmax[(i + mo) * 8 + wave] = rmax[i];
  }
  __syncthreads();
#pragma unroll
  for (int i = 0; i < 8; ++i) {
    float mx = pmax[(i + mo) * 8];
#pragma unroll
    for (int w = 1; w < 8; ++w) mx = fmaxf(mx, pmax[(i + mo) * 8 + w]);
    rmax[i] = mx;
  }
#pragma unroll
  for (int i = 0; i < 8; ++i) {
    float ssum = 0.f;
#pragma unroll
    for (int j = 0; j < 8; ++j) {
      float e = __expf(acc[j][i] - rmax[i]);
      acc[j][i] = e;
      ssum += e;
    }
    rsum[i] = hsum16(ssum);
  }
  if ((lane & 15) == 0) {
#pragma unroll
    for (int i = 0; i < 8; ++i) psum[(i + mo) * 8 + wave] = rsum[i];
  }
  __syncthreads();
#pragma unroll
  for (int i = 0; i < 8; ++i) {
    float sm = 0.f;
#pragma unroll
    for (int w = 0; w < 8; ++w) sm += psum[(i + mo) * 8 + w];
    rsum[i] = 1.0f / sm;
  }
#pragma unroll
  for (int j = 0; j < 8; ++j) {
    const int s = (wave + j * 8) * 16 + (lane & 15);
#pragma unroll
    for (int i = 0; i < 8; ++i)
      pr[(i + mo) * 1024 + s] = (__bf16)(acc[j][i] * rsum[i]);
  }
  __syncthreads();

  // phase 3: attn = probs @ V (B-fragments from transposed V, contiguous)
  const __bf16* Vb = Vt + (long)b * H_ * S_;
#pragma unroll
  for (int t = 0; t < 2; ++t) {
    const int ht = wave + t * 8;
    v8f a = zero;
    for (int sb = 0; sb < S_; sb += 32) {
      v16bf af = load_fragA(pr, 1024, sb);
      v16bf bf = load_fragB(Vb, S_, ht * 16, sb);
      a = wmma_bf16(af, bf, a);
    }
    const int n = ht * 16 + (lane & 15);
#pragma unroll
    for (int i = 0; i < 8; ++i)
      cat4[(long)(b * P_ + p0 + i + mo) * 1024 + 768 + n] = (__bf16)a[i];
  }
}

// ---------------------------------------------------------------- final fc dot
__global__ void li_kernel(const __bf16* __restrict__ cat4, const __bf16* __restrict__ mid,
                          const float* __restrict__ fcW, const float* __restrict__ fcb,
                          float* __restrict__ li, int Nrows) {
  const int wave = threadIdx.x >> 5, lane = threadIdx.x & 31;
  const int row = blockIdx.x * 8 + wave;
  if (row >= Nrows) return;
  const __bf16* c = cat4 + (long)row * 1024;
  const __bf16* m = mid + (long)row * 256;
  float s = 0.f;
  for (int j = lane; j < 1024; j += 32) s += (float)c[j] * fcW[j];
  for (int j = lane; j < 256;  j += 32) s += (float)m[j] * fcW[1024 + j];
  s = wsum32(s);
  if (lane == 0) li[row] = s + fcb[0];
}

// ---------------------------------------------------------------- log-softmax over P
__global__ void logsoftmax_kernel(const float* __restrict__ li, float* __restrict__ out) {
  __shared__ float redm[8];
  __shared__ float reds[8];
  const int b = blockIdx.x;
  const int wave = threadIdx.x >> 5, lane = threadIdx.x & 31;
  const float* x = li + (long)b * P_;
  float mx = -3.4e38f;
  for (int j = threadIdx.x; j < P_; j += 256) mx = fmaxf(mx, x[j]);
  mx = wmax32(mx);
  if (lane == 0) redm[wave] = mx;
  __syncthreads();
  float bm = redm[0];
#pragma unroll
  for (int i = 1; i < 8; ++i) bm = fmaxf(bm, redm[i]);
  float s = 0.f;
  for (int j = threadIdx.x; j < P_; j += 256) s += __expf(x[j] - bm);
  s = wsum32(s);
  if (lane == 0) reds[wave] = s;
  __syncthreads();
  float bs = 0.f;
#pragma unroll
  for (int i = 0; i < 8; ++i) bs += reds[i];
  const float lse = bm + logf(bs);
  for (int j = threadIdx.x; j < P_; j += 256) out[(long)b * P_ + j] = x[j] - lse;
}

// ---------------------------------------------------------------- launcher
extern "C" void kernel_launch(void* const* d_in, const int* in_sizes, int n_in,
                              void* d_out, int out_size, void* d_ws, size_t ws_size,
                              hipStream_t stream) {
  (void)in_sizes; (void)n_in; (void)out_size; (void)ws_size;

  const float* goals  = (const float*)d_in[0];
  const float* hidden = (const float*)d_in[1];
  const float* mask   = (const float*)d_in[2];
  const float *m0W=(const float*)d_in[3],  *m0b=(const float*)d_in[4],
              *m0g=(const float*)d_in[5],  *m0be=(const float*)d_in[6];
  const float *m1W=(const float*)d_in[7],  *m1b=(const float*)d_in[8],
              *m1g=(const float*)d_in[9],  *m1be=(const float*)d_in[10];
  const float *m2W=(const float*)d_in[11], *m2b=(const float*)d_in[12],
              *m2g=(const float*)d_in[13], *m2be=(const float*)d_in[14];
  const float *p0W=(const float*)d_in[15], *p0b=(const float*)d_in[16],
              *p0g=(const float*)d_in[17], *p0be=(const float*)d_in[18];
  const float *p1W=(const float*)d_in[19], *p1b=(const float*)d_in[20],
              *p1g=(const float*)d_in[21], *p1be=(const float*)d_in[22];
  const float *p2W=(const float*)d_in[23], *p2b=(const float*)d_in[24],
              *p2g=(const float*)d_in[25], *p2be=(const float*)d_in[26];
  const float *qW=(const float*)d_in[27],  *qb=(const float*)d_in[28];
  const float *kW=(const float*)d_in[29],  *kbias=(const float*)d_in[30];
  const float *vW=(const float*)d_in[31],  *vbias=(const float*)d_in[32];
  const float *dmW=(const float*)d_in[33], *dmb=(const float*)d_in[34],
              *dmg=(const float*)d_in[35], *dmbe=(const float*)d_in[36];
  const float *fcW=(const float*)d_in[37], *fcb=(const float*)d_in[38];

  char* ws = (char*)d_ws;
  size_t off = 0;
  auto alloc = [&](size_t bytes) -> void* {
    void* p = ws + off;
    off += (bytes + 255) & ~(size_t)255;
    return p;
  };

  __bf16* hidB  = (__bf16*)alloc((size_t)NS_ * H_ * 2);
  __bf16* Wm1   = (__bf16*)alloc((size_t)256 * 256 * 2);
  __bf16* Wm2   = (__bf16*)alloc((size_t)256 * 256 * 2);
  __bf16* Wp1   = (__bf16*)alloc((size_t)128 * 256 * 2);
  __bf16* Wp2   = (__bf16*)alloc((size_t)256 * 256 * 2);
  __bf16* Wq    = (__bf16*)alloc((size_t)256 * 256 * 2);
  __bf16* Wk    = (__bf16*)alloc((size_t)256 * 256 * 2);
  __bf16* Wv    = (__bf16*)alloc((size_t)256 * 256 * 2);
  __bf16* Wdm   = (__bf16*)alloc((size_t)256 * 1024 * 2);
  __bf16* feat0 = (__bf16*)alloc((size_t)NP_ * 256 * 2);
  __bf16* feat1 = (__bf16*)alloc((size_t)NP_ * 256 * 2);
  __bf16* x1    = (__bf16*)alloc((size_t)NP_ * 256 * 2);
  __bf16* x2    = (__bf16*)alloc((size_t)NP_ * 256 * 2);
  __bf16* qbuf  = (__bf16*)alloc((size_t)NP_ * 256 * 2);
  __bf16* kbuf  = (__bf16*)alloc((size_t)NS_ * 256 * 2);
  __bf16* vtbuf = (__bf16*)alloc((size_t)B_ * 256 * S_ * 2);
  __bf16* cat4  = (__bf16*)alloc((size_t)NP_ * 1024 * 2);
  __bf16* mid   = (__bf16*)alloc((size_t)NP_ * 256 * 2);
  float*  libuf = (float*) alloc((size_t)NP_ * 4);

  // casts
  cast_f32_bf16<<<(NS_ * H_ + 255) / 256, 256, 0, stream>>>(hidden, hidB, NS_ * H_);
  cast_f32_bf16<<<(256 * 256 + 255) / 256, 256, 0, stream>>>(m1W, Wm1, 256 * 256);
  cast_f32_bf16<<<(256 * 256 + 255) / 256, 256, 0, stream>>>(m2W, Wm2, 256 * 256);
  cast_f32_bf16<<<(128 * 256 + 255) / 256, 256, 0, stream>>>(p1W, Wp1, 128 * 256);
  cast_f32_bf16<<<(256 * 256 + 255) / 256, 256, 0, stream>>>(p2W, Wp2, 256 * 256);
  cast_f32_bf16<<<(256 * 256 + 255) / 256, 256, 0, stream>>>(qW, Wq, 256 * 256);
  cast_f32_bf16<<<(256 * 256 + 255) / 256, 256, 0, stream>>>(kW, Wk, 256 * 256);
  cast_f32_bf16<<<(256 * 256 + 255) / 256, 256, 0, stream>>>(vW, Wv, 256 * 256);
  cast_f32_bf16<<<(256 * 1024 + 255) / 256, 256, 0, stream>>>(dmW, Wdm, 256 * 1024);

  // K=2 heads
  head_mlp<<<NP_ / 8, 256, 0, stream>>>(goals, m0W, m0b, m0g, m0be, feat0, 256, 256, NP_);
  head_mlp<<<NP_ / 8, 256, 0, stream>>>(goals, p0W, p0b, p0g, p0be, x1, 256, 128, NP_);
  fill_agent<<<NP_ * H_ / 256, 256, 0, stream>>>(hidden, cat4, x1, x2);

  // feat chain -> cat4[:,512:768]
  gemm_bf16<2,1><<<NP_ / 16, 256, 0, stream>>>(feat0, 256, Wm1, 256, m1b, m1g, m1be,
                                               feat1, 256, 256, 256, 1.0f, 0);
  gemm_bf16<2,1><<<NP_ / 16, 256, 0, stream>>>(feat1, 256, Wm2, 256, m2b, m2g, m2be,
                                               cat4 + 512, 1024, 256, 256, 1.0f, 0);
  // sub chain -> cat4[:,256:512]
  gemm_bf16<1,1><<<NP_ / 16, 256, 0, stream>>>(x1, 256, Wp1, 256, p1b, p1g, p1be,
                                               x2, 256, 256, 128, 1.0f, 0);
  gemm_bf16<2,1><<<NP_ / 16, 256, 0, stream>>>(x2, 256, Wp2, 256, p2b, p2g, p2be,
                                               cat4 + 256, 1024, 256, 256, 1.0f, 0);
  // q (scaled by 1/sqrt(H)), k, v(transposed)
  gemm_bf16<2,0><<<NP_ / 16, 256, 0, stream>>>(cat4 + 512, 1024, Wq, 256, qb, nullptr, nullptr,
                                               qbuf, 256, 256, 256, 0.0625f, 0);
  gemm_bf16<2,0><<<NS_ / 16, 256, 0, stream>>>(hidB, 256, Wk, 256, kbias, nullptr, nullptr,
                                               kbuf, 256, 256, 256, 1.0f, 0);
  gemm_bf16<2,2><<<NS_ / 16, 256, 0, stream>>>(hidB, 256, Wv, 256, vbias, nullptr, nullptr,
                                               vtbuf, 0, 256, 256, 1.0f, S_);

  // attention -> cat4[:,768:1024]
  attention_kernel<<<B_ * (P_ / 16), 256, 0, stream>>>(qbuf, kbuf, vtbuf, mask, cat4);

  // dm MLP on cat4 (K=1024)
  gemm_bf16<2,1><<<NP_ / 16, 256, 0, stream>>>(cat4, 1024, Wdm, 1024, dmb, dmg, dmbe,
                                               mid, 256, 1024, 256, 1.0f, 0);

  // fc dot + log-softmax
  li_kernel<<<NP_ / 8, 256, 0, stream>>>(cat4, mid, fcW, fcb, libuf, NP_);
  logsoftmax_kernel<<<B_, 256, 0, stream>>>(libuf, (float*)d_out);
}
